// DPGN_74174085202585
// MI455X (gfx1250) — matrix operand order repaired and checked by reading, not compiled, not run on hardware
//
#include <hip/hip_runtime.h>
#include <math.h>

typedef __attribute__((ext_vector_type(16))) __bf16 v16bf;
typedef __attribute__((ext_vector_type(8)))  __bf16 v8bf;
typedef __attribute__((ext_vector_type(8)))  float  v8f;

#define GG 2
#define BB 256
#define NN 30
#define SS 25
#define DD 128
#define LRELU_NEG 0.01f
#define BN_SCALE 0.999995000037499687f   // 1/sqrt(1+1e-5)
#define NPAIR (NN*NN)                    // 900

// ---------------- WMMA helpers (CDNA5 layouts, CDNA5 ISA §7.12.2) ----

__device__ inline v8f wmma_bf16(v16bf a, v16bf b, v8f c) {
  return __builtin_amdgcn_wmma_f32_16x16x32_bf16(
      /*neg_a=*/false, a, /*neg_b=*/false, b,
      /*c_mod=*/(short)0, c, /*reuse_a=*/false, /*reuse_b=*/false);
}

__device__ inline v16bf make_v16(v8bf lo, v8bf hi) {
  v16bf r;
#pragma unroll
  for (int e = 0; e < 8; ++e) { r[e] = lo[e]; r[e + 8] = hi[e]; }
  return r;
}

// A-tile (16 rows x 32 k) bf16 from row-major LDS tile [16][ld].
// Lane l: row = l&15; lanes 0-15 take K = kbase+0..7 (regs 0-3) and kbase+16..23
// (regs 4-7); lanes 16-31 take kbase+8..15 and kbase+24..31.
__device__ inline v16bf load_a_tile(const __bf16* base, int ld, int kbase) {
  const int lane = threadIdx.x & 31;
  const int row = lane & 15;
  const int khalf = (lane >> 4) ? 8 : 0;
  const __bf16* p = base + row * ld + kbase + khalf;
  v8bf lo = *(const v8bf*)(p);
  v8bf hi = *(const v8bf*)(p + 16);
  return make_v16(lo, hi);
}

// B-tile (32 k x 16 n) from row-major bf16 weight W[outC][inC]; B(k,n)=W[n][k].
// Lane l: n = l&15; lanes 0-15 hold K = kt*32+0..7 and +16..23, lanes 16-31 K = kt*32+8..15
// and +24..31 (same K order as the A tile).
__device__ inline v16bf load_b_tile(const __bf16* w, int inC, int nt, int kt) {
  const int lane = threadIdx.x & 31;
  const int n = nt * 16 + (lane & 15);
  const int k = kt * 32 + ((lane >> 4) ? 8 : 0);
  const __bf16* p = w + n * inC + k;
  v8bf lo = *(const v8bf*)(p);
  v8bf hi = *(const v8bf*)(p + 16);
  return make_v16(lo, hi);
}

__device__ inline float lrelu(float x) { return (x >= 0.f) ? x : LRELU_NEG * x; }

// Edge update for one row i: e = l1n(s*em)*esum + eye + 1e-6; e /= rowsum.
__device__ inline void edge_row_update(int i, const float* __restrict__ er,
                                       const float* sr, float* __restrict__ orow) {
  float em[NN];
  float esum = 0.f, tsum = 0.f;
#pragma unroll 1
  for (int j = 0; j < NN; ++j) {
    float e = (j == i) ? 0.f : er[j];
    em[j] = e;
    esum += e;
    tsum += fabsf(sr[j] * e);
  }
  float inv = esum / fmaxf(tsum, 1e-12f);
  float row[NN];
  float rsum = 0.f;
#pragma unroll 1
  for (int j = 0; j < NN; ++j) {
    float ev = sr[j] * em[j] * inv + ((j == i) ? 1.f : 0.f) + 1e-6f;
    row[j] = ev;
    rsum += ev;
  }
  float rinv = 1.f / rsum;
#pragma unroll 1
  for (int j = 0; j < NN; ++j) orow[j] = row[j] * rinv;
}

// ---------------- prep: copy states, convert weights to bf16 --------------------

__global__ void prep_kernel(const float* __restrict__ pn_in,
                            const float* __restrict__ dn_in,
                            const float* __restrict__ ps_w1,
                            const float* __restrict__ ps_w2,
                            const float* __restrict__ dp_w1,
                            const float* __restrict__ dp_w2,
                            float* __restrict__ pn_st, float* __restrict__ dn_st,
                            __bf16* __restrict__ bw_ps1, __bf16* __restrict__ bw_ps2,
                            __bf16* __restrict__ bw_dp1, __bf16* __restrict__ bw_dp2) {
  const int idx = blockIdx.x * blockDim.x + threadIdx.x;
  const int stride = gridDim.x * blockDim.x;
  for (int i = idx; i < BB * NN * DD; i += stride) pn_st[i] = pn_in[i];
  for (int i = idx; i < BB * NN * SS; i += stride) dn_st[i] = dn_in[i];
  for (int i = idx; i < GG * 2 * DD * DD; i += stride) bw_ps1[i] = (__bf16)ps_w1[i];
  for (int i = idx; i < GG * DD * 2 * DD; i += stride) bw_ps2[i] = (__bf16)ps_w2[i];
  for (int i = idx; i < GG * 2 * DD * 2 * DD; i += stride) bw_dp1[i] = (__bf16)dp_w1[i];
  for (int i = idx; i < GG * DD * 2 * DD; i += stride) bw_dp2[i] = (__bf16)dp_w2[i];
}

// ---------------- point similarity: WMMA MLP over 900 pair-rows ----------------

__global__ __launch_bounds__(256) void sim_point_kernel(
    const float* __restrict__ pn,      // [B][30][128]
    const float* __restrict__ e_last,  // [B][30][30]
    const __bf16* __restrict__ w1,     // [256][128] bf16
    const __bf16* __restrict__ w2,     // [128][256] bf16
    const float* __restrict__ g1, const float* __restrict__ b1,
    const float* __restrict__ g2, const float* __restrict__ b2,
    const float* __restrict__ w3, const float* __restrict__ b3,
    float* __restrict__ out_edge,      // [B][30][30]
    float* __restrict__ out_l2) {      // [B][30][30]
  __shared__ float vsh[NN][DD];     // 15360 B
  __shared__ float smat[NPAIR];     // 3600 B
  __shared__ __bf16 aT[16][DD];     // 4096 B
  __shared__ __bf16 h1[16][2 * DD]; // 8192 B
  __shared__ float h2f[16][DD];     // 8192 B
  __shared__ float rpart[16][16];   // 1024 B  (node_l2 partials)
  __shared__ float wpart[16][16];   // 1024 B  (w3-dot partials)
  __shared__ float w3s[DD];         // 512 B

  const int b = blockIdx.x;
  const int t = threadIdx.x;
  const int wave = t >> 5, lane = t & 31;

  for (int idx = t; idx < NN * DD; idx += 256) vsh[idx / DD][idx % DD] = pn[b * NN * DD + idx];
  for (int idx = t; idx < DD; idx += 256) w3s[idx] = w3[idx];

  // Hoist B-tiles (reused across all 57 M-tiles) and per-lane BN params.
  const int nt0 = 2 * wave;
  v16bf B1[2][4];
#pragma unroll
  for (int j = 0; j < 2; ++j)
#pragma unroll
    for (int k = 0; k < 4; ++k) B1[j][k] = load_b_tile(w1, DD, nt0 + j, k);
  v16bf B2[8];
#pragma unroll
  for (int k = 0; k < 8; ++k) B2[k] = load_b_tile(w2, 2 * DD, wave, k);

  float sc1[2], bi1[2];
#pragma unroll
  for (int j = 0; j < 2; ++j) {
    int o = (nt0 + j) * 16 + (lane & 15);
    sc1[j] = g1[o] * BN_SCALE;
    bi1[j] = b1[o];
  }
  const int o2 = wave * 16 + (lane & 15);
  const float sc2 = g2[o2] * BN_SCALE, bi2 = b2[o2];
  const float b3v = b3[0];
  __syncthreads();

  const int rrow = t >> 4, rsub = t & 15;  // 16 rows x 16 threads layout

  for (int m0 = 0; m0 < NPAIR; m0 += 16) {
    // build 16x128 squared-difference tile (bf16) + per-row sums for node_l2
    {
      const int p = m0 + rrow;
      int i = 0, j = 0;
      if (p < NPAIR) { i = p / NN; j = p % NN; }
      float partial = 0.f;
#pragma unroll
      for (int kk = 0; kk < 8; ++kk) {
        int c = rsub + kk * 16;
        float diff = vsh[i][c] - vsh[j][c];
        float dsq = (p < NPAIR) ? diff * diff : 0.f;
        aT[rrow][c] = (__bf16)dsq;
        partial += dsq;
      }
      rpart[rrow][rsub] = partial;
    }
    __syncthreads();
    if (t < 16) {
      int p = m0 + t;
      if (p < NPAIR) {
        float s = 0.f;
#pragma unroll
        for (int k = 0; k < 16; ++k) s += rpart[t][k];
        out_l2[b * NPAIR + p] = -s;
      }
    }
    // GEMM1: [16x128] @ W1^T -> [16x256], BN + lrelu, to bf16 LDS
#pragma unroll
    for (int j = 0; j < 2; ++j) {
      v8f acc = {};
#pragma unroll
      for (int k = 0; k < 4; ++k) {
        v16bf a = load_a_tile(&aT[0][0], DD, k * 32);
        acc = wmma_bf16(a, B1[j][k], acc);
      }
      const int o = (nt0 + j) * 16 + (lane & 15);
      const int rbase = (lane >> 4) ? 8 : 0;
#pragma unroll
      for (int q = 0; q < 8; ++q)
        h1[rbase + q][o] = (__bf16)lrelu(acc[q] * sc1[j] + bi1[j]);
    }
    __syncthreads();
    // GEMM2: [16x256] @ W2^T -> [16x128], BN + lrelu, to f32 LDS
    {
      v8f acc = {};
#pragma unroll
      for (int k = 0; k < 8; ++k) {
        v16bf a = load_a_tile(&h1[0][0], 2 * DD, k * 32);
        acc = wmma_bf16(a, B2[k], acc);
      }
      const int rbase = (lane >> 4) ? 8 : 0;
#pragma unroll
      for (int q = 0; q < 8; ++q)
        h2f[rbase + q][o2] = lrelu(acc[q] * sc2 + bi2);
    }
    __syncthreads();
    // w3 dot: all 256 threads compute 8-channel partials (contiguous 32B reads)
    {
      float partial = 0.f;
#pragma unroll
      for (int kk = 0; kk < 8; ++kk) {
        int c = rsub * 8 + kk;
        partial += h2f[rrow][c] * w3s[c];
      }
      wpart[rrow][rsub] = partial;
    }
    __syncthreads();
    if (t < 16) {
      int p = m0 + t;
      if (p < NPAIR) {
        float s = b3v;
#pragma unroll
        for (int k = 0; k < 16; ++k) s += wpart[t][k];
        smat[p] = 1.f / (1.f + expf(-s));
      }
    }
    __syncthreads();
  }

  // edge normalization update
  if (t < NN) {
    edge_row_update(t, e_last + b * NPAIR + t * NN, &smat[t * NN],
                    out_edge + b * NPAIR + t * NN);
  }
}

// ---------------- P2DAgg: dn = lrelu([pe[:, :S], dn] @ p2d_w^T + b) -------------

__global__ __launch_bounds__(256) void p2d_kernel(const float* __restrict__ pe,
                                                  float* __restrict__ dn,
                                                  const float* __restrict__ w,
                                                  const float* __restrict__ bias) {
  __shared__ float cat[NN][2 * SS];
  __shared__ float wsh[SS][2 * SS];
  const int b = blockIdx.x, t = threadIdx.x;
  for (int idx = t; idx < NN * SS; idx += 256) {
    int i = idx / SS, k = idx % SS;
    cat[i][k] = pe[b * NPAIR + i * NN + k];
    cat[i][SS + k] = dn[b * NN * SS + idx];
  }
  for (int idx = t; idx < SS * 2 * SS; idx += 256) wsh[idx / (2 * SS)][idx % (2 * SS)] = w[idx];
  __syncthreads();
  for (int idx = t; idx < NN * SS; idx += 256) {
    int i = idx / SS, o = idx % SS;
    float s = bias[o];
#pragma unroll 1
    for (int k = 0; k < 2 * SS; ++k) s += cat[i][k] * wsh[o][k];
    dn[b * NN * SS + idx] = lrelu(s);
  }
}

// ---------------- distribution similarity (K=25/50: VALU) ----------------------

__global__ __launch_bounds__(256) void sim_dist_kernel(
    const float* __restrict__ dn,      // [B][30][25]
    const float* __restrict__ e_last,  // [B][30][30]
    const float* __restrict__ w1,      // [50][25]
    const float* __restrict__ g1, const float* __restrict__ b1,
    const float* __restrict__ w2,      // [25][50]
    const float* __restrict__ g2, const float* __restrict__ b2,
    const float* __restrict__ w3, const float* __restrict__ b3,
    float* __restrict__ out_edge) {
  __shared__ float dv[NN][SS];
  __shared__ float smat[NPAIR];
  __shared__ float w1s[2 * SS][SS];
  __shared__ float w2s[SS][2 * SS];
  __shared__ float g1s[2 * SS], b1s[2 * SS], g2s[SS], b2s[SS], w3s[SS];
  const int b = blockIdx.x, t = threadIdx.x;
  for (int idx = t; idx < NN * SS; idx += 256) dv[idx / SS][idx % SS] = dn[b * NN * SS + idx];
  for (int idx = t; idx < 2 * SS * SS; idx += 256) w1s[idx / SS][idx % SS] = w1[idx];
  for (int idx = t; idx < SS * 2 * SS; idx += 256) w2s[idx / (2 * SS)][idx % (2 * SS)] = w2[idx];
  if (t < 2 * SS) { g1s[t] = g1[t] * BN_SCALE; b1s[t] = b1[t]; }
  if (t < SS) { g2s[t] = g2[t] * BN_SCALE; b2s[t] = b2[t]; w3s[t] = w3[t]; }
  __syncthreads();
  const float b3v = b3[0];
  for (int p = t; p < NPAIR; p += 256) {
    const int i = p / NN, j = p % NN;
    float d[SS];
#pragma unroll
    for (int c = 0; c < SS; ++c) {
      float df = dv[i][c] - dv[j][c];
      d[c] = df * df;
    }
    float h1v[2 * SS];
#pragma unroll 1
    for (int o = 0; o < 2 * SS; ++o) {
      float s = 0.f;
#pragma unroll
      for (int c = 0; c < SS; ++c) s += d[c] * w1s[o][c];
      h1v[o] = lrelu(s * g1s[o] + b1s[o]);
    }
    float sdot = b3v;
#pragma unroll 1
    for (int o = 0; o < SS; ++o) {
      float s = 0.f;
#pragma unroll
      for (int c = 0; c < 2 * SS; ++c) s += h1v[c] * w2s[o][c];
      sdot += lrelu(s * g2s[o] + b2s[o]) * w3s[o];
    }
    smat[p] = 1.f / (1.f + expf(-sdot));
  }
  __syncthreads();
  if (t < NN) {
    edge_row_update(t, e_last + b * NPAIR + t * NN, &smat[t * NN],
                    out_edge + b * NPAIR + t * NN);
  }
}

// ---------------- D2PAgg: aggregate + 2-layer WMMA MLP -------------------------

__global__ __launch_bounds__(256) void d2p_kernel(
    const float* __restrict__ de,     // [B][30][30] (dist edge out)
    float* __restrict__ pn,           // state [B][30][128], updated in place
    const __bf16* __restrict__ w1,    // [256][256] bf16
    const float* __restrict__ g1, const float* __restrict__ b1,
    const __bf16* __restrict__ w2,    // [128][256] bf16
    const float* __restrict__ g2, const float* __restrict__ b2) {
  __shared__ float vsh[NN][DD];       // 15360 B
  __shared__ float ef[NN][NN];        // 3600 B
  __shared__ __bf16 nf[32][2 * DD];   // 16384 B
  __shared__ __bf16 hb[32][2 * DD];   // 16384 B
  const int b = blockIdx.x, t = threadIdx.x;
  const int wave = t >> 5, lane = t & 31;

  for (int idx = t; idx < NN * DD; idx += 256) vsh[idx / DD][idx % DD] = pn[b * NN * DD + idx];
  for (int idx = t; idx < 2 * (2 * DD); idx += 256)
    nf[NN + idx / (2 * DD)][idx % (2 * DD)] = (__bf16)0.f;   // pad rows 30,31
  if (t < NN) {
    const int i = t;
    float row[NN];
    float asum = 0.f;
#pragma unroll 1
    for (int j = 0; j < NN; ++j) {
      float e = (j == i) ? 0.f : de[b * NPAIR + i * NN + j];
      row[j] = e;
      asum += fabsf(e);
    }
    float inv = 1.f / fmaxf(asum, 1e-12f);
#pragma unroll 1
    for (int j = 0; j < NN; ++j) ef[i][j] = row[j] * inv;
  }
  __syncthreads();
  // aggr = ef @ v ; nf = [v, aggr] in bf16
  for (int idx = t; idx < NN * DD; idx += 256) {
    const int i = idx / DD, c = idx % DD;
    float a = 0.f;
#pragma unroll 1
    for (int j = 0; j < NN; ++j) a += ef[i][j] * vsh[j][c];
    nf[i][c] = (__bf16)vsh[i][c];
    nf[i][DD + c] = (__bf16)a;
  }
  __syncthreads();
  // GEMM1: [32x256] @ W1^T -> [32x256]
  {
    const int nt = 2 * wave;
    v16bf Bt[2][8];
#pragma unroll
    for (int j = 0; j < 2; ++j)
#pragma unroll
      for (int k = 0; k < 8; ++k) Bt[j][k] = load_b_tile(w1, 2 * DD, nt + j, k);
    float sc[2], bi[2];
#pragma unroll
    for (int j = 0; j < 2; ++j) {
      int o = (nt + j) * 16 + (lane & 15);
      sc[j] = g1[o] * BN_SCALE;
      bi[j] = b1[o];
    }
#pragma unroll
    for (int mt = 0; mt < 2; ++mt)
#pragma unroll
      for (int j = 0; j < 2; ++j) {
        v8f acc = {};
#pragma unroll
        for (int k = 0; k < 8; ++k) {
          v16bf a = load_a_tile(&nf[mt * 16][0], 2 * DD, k * 32);
          acc = wmma_bf16(a, Bt[j][k], acc);
        }
        const int o = (nt + j) * 16 + (lane & 15);
        const int rbase = mt * 16 + ((lane >> 4) ? 8 : 0);
#pragma unroll
        for (int q = 0; q < 8; ++q)
          hb[rbase + q][o] = (__bf16)lrelu(acc[q] * sc[j] + bi[j]);
      }
  }
  __syncthreads();
  // GEMM2: [32x256] @ W2^T -> [30x128] (rows >= 30 masked)
  {
    const int nt = wave;
    v16bf Bt[8];
#pragma unroll
    for (int k = 0; k < 8; ++k) Bt[k] = load_b_tile(w2, 2 * DD, nt, k);
    const int o = nt * 16 + (lane & 15);
    const float sc = g2[o] * BN_SCALE, bi = b2[o];
#pragma unroll
    for (int mt = 0; mt < 2; ++mt) {
      v8f acc = {};
#pragma unroll
      for (int k = 0; k < 8; ++k) {
        v16bf a = load_a_tile(&hb[mt * 16][0], 2 * DD, k * 32);
        acc = wmma_bf16(a, Bt[k], acc);
      }
      const int rbase = mt * 16 + ((lane >> 4) ? 8 : 0);
#pragma unroll
      for (int q = 0; q < 8; ++q) {
        const int row = rbase + q;
        if (row < NN) pn[b * NN * DD + row * DD + o] = lrelu(acc[q] * sc + bi);
      }
    }
  }
}

// ---------------- launch --------------------------------------------------------

extern "C" void kernel_launch(void* const* d_in, const int* in_sizes, int n_in,
                              void* d_out, int out_size, void* d_ws, size_t ws_size,
                              hipStream_t stream) {
  (void)in_sizes; (void)n_in; (void)out_size; (void)ws_size;
  const float* pn_in = (const float*)d_in[0];
  const float* pe_in = (const float*)d_in[1];
  const float* dn_in = (const float*)d_in[2];
  const float* de_in = (const float*)d_in[3];
  const float* ps_w1 = (const float*)d_in[4];
  const float* ps_g1 = (const float*)d_in[5];
  const float* ps_b1 = (const float*)d_in[6];
  const float* ps_w2 = (const float*)d_in[7];
  const float* ps_g2 = (const float*)d_in[8];
  const float* ps_b2 = (const float*)d_in[9];
  const float* ps_w3 = (const float*)d_in[10];
  const float* ps_b3 = (const float*)d_in[11];
  const float* p2d_w = (const float*)d_in[12];
  const float* p2d_b = (const float*)d_in[13];
  const float* ds_w1 = (const float*)d_in[14];
  const float* ds_g1 = (const float*)d_in[15];
  const float* ds_b1 = (const float*)d_in[16];
  const float* ds_w2 = (const float*)d_in[17];
  const float* ds_g2 = (const float*)d_in[18];
  const float* ds_b2 = (const float*)d_in[19];
  const float* ds_w3 = (const float*)d_in[20];
  const float* ds_b3 = (const float*)d_in[21];
  const float* dp_w1 = (const float*)d_in[22];
  const float* dp_g1 = (const float*)d_in[23];
  const float* dp_b1 = (const float*)d_in[24];
  const float* dp_w2 = (const float*)d_in[25];
  const float* dp_g2 = (const float*)d_in[26];
  const float* dp_b2 = (const float*)d_in[27];

  char* ws = (char*)d_ws;
  size_t off = 0;
  float* pn_st = (float*)(ws + off); off += (size_t)BB * NN * DD * 4;
  float* dn_st = (float*)(ws + off); off += (size_t)BB * NN * SS * 4;
  off = (off + 255) & ~(size_t)255;
  __bf16* bw_ps1 = (__bf16*)(ws + off); off += (size_t)GG * 2 * DD * DD * 2;
  __bf16* bw_ps2 = (__bf16*)(ws + off); off += (size_t)GG * DD * 2 * DD * 2;
  __bf16* bw_dp1 = (__bf16*)(ws + off); off += (size_t)GG * 2 * DD * 2 * DD * 2;
  __bf16* bw_dp2 = (__bf16*)(ws + off); off += (size_t)GG * DD * 2 * DD * 2;

  float* out = (float*)d_out;
  auto out_slice = [&](int g, int k) {
    return out + ((size_t)(g * 3 + k)) * BB * NPAIR;
  };

  prep_kernel<<<256, 256, 0, stream>>>(pn_in, dn_in, ps_w1, ps_w2, dp_w1, dp_w2,
                                       pn_st, dn_st, bw_ps1, bw_ps2, bw_dp1, bw_dp2);

  for (int g = 0; g < GG; ++g) {
    const float* pe_last = (g == 0) ? pe_in : out_slice(g - 1, 0);
    const float* de_last = (g == 0) ? de_in : out_slice(g - 1, 2);

    sim_point_kernel<<<BB, 256, 0, stream>>>(
        pn_st, pe_last,
        bw_ps1 + (size_t)g * 2 * DD * DD, bw_ps2 + (size_t)g * DD * 2 * DD,
        ps_g1 + g * 2 * DD, ps_b1 + g * 2 * DD, ps_g2 + g * DD, ps_b2 + g * DD,
        ps_w3 + g * DD, ps_b3 + g, out_slice(g, 0), out_slice(g, 1));

    p2d_kernel<<<BB, 256, 0, stream>>>(out_slice(g, 0), dn_st,
                                       p2d_w + g * SS * 2 * SS, p2d_b + g * SS);

    sim_dist_kernel<<<BB, 256, 0, stream>>>(
        dn_st, de_last,
        ds_w1 + g * 2 * SS * SS, ds_g1 + g * 2 * SS, ds_b1 + g * 2 * SS,
        ds_w2 + g * SS * 2 * SS, ds_g2 + g * SS, ds_b2 + g * SS,
        ds_w3 + g * SS, ds_b3 + g, out_slice(g, 2));

    d2p_kernel<<<BB, 256, 0, stream>>>(
        out_slice(g, 2), pn_st,
        bw_dp1 + (size_t)g * 2 * DD * 2 * DD, dp_g1 + g * 2 * DD, dp_b1 + g * 2 * DD,
        bw_dp2 + (size_t)g * DD * 2 * DD, dp_g2 + g * DD, dp_b2 + g * DD);
  }
}
